// AtomEncoder_37211596653156
// MI455X (gfx1250) — matrix operand order repaired and checked by reading, not compiled, not run on hardware
//
#include <hip/hip_runtime.h>
#include <math.h>

#define N_ATOMS 100000
#define N_EDGES 800000
#define ETOT    (N_EDGES + N_ATOMS)
#define EDIM    13
#define INF_    28
#define TT      10
#define CH      128          // H * fout for every GAT / linear output width

typedef _Float16 f16;
typedef f16   v16h __attribute__((ext_vector_type(16)));
typedef float v8f  __attribute__((ext_vector_type(8)));

static inline int cdiv(int a, int b) { return (a + b - 1) / b; }

// ---------------------------------------------------------------- utilities
__global__ void k_fill_f32(float* __restrict__ p, float v, int n) {
  int i = blockIdx.x * blockDim.x + threadIdx.x;
  if (i < n) p[i] = v;
}
__global__ void k_fill_i32(int* __restrict__ p, int v, int n) {
  int i = blockIdx.x * blockDim.x + threadIdx.x;
  if (i < n) p[i] = v;
}

__device__ inline void atomicMaxF(float* addr, float val) {
  // ordered-int trick; addr must be initialized to -inf
  if (val >= 0.f) atomicMax((int*)addr, __float_as_int(val));
  else            atomicMin((unsigned int*)addr, (unsigned int)__float_as_int(val));
}

// --------------------------------------------------- weight convert+transpose
// W:  [Tn, K, CH] f32 row-major   ->   Wt: [Tn, CH, KP] f16 (column-major, K padded)
__global__ void k_prep_w(const float* __restrict__ W, f16* __restrict__ Wt,
                         int K, int KP, int Tn) {
  int total = Tn * CH * KP;
  for (int i = blockIdx.x * blockDim.x + threadIdx.x; i < total;
       i += gridDim.x * blockDim.x) {
    int kp = i % KP;
    int rest = i / KP;
    int n = rest % CH;
    int t = rest / CH;
    float v = (kp < K) ? W[((size_t)t * K + kp) * CH + n] : 0.f;
    Wt[i] = (f16)v;
  }
}

// ------------------------------------------------------------ self-loop attr
__global__ void k_loop_accum(const int* __restrict__ ei, const float* __restrict__ ea,
                             float* __restrict__ deg, float* __restrict__ la) {
  int i = blockIdx.x * blockDim.x + threadIdx.x;
  if (i >= N_EDGES * EDIM) return;
  int e = i / EDIM, j = i % EDIM;
  int d = ei[N_EDGES + e];
  atomicAdd(&la[(size_t)d * EDIM + j], ea[(size_t)e * EDIM + j]);
  if (j == 0) atomicAdd(&deg[d], 1.f);
}
__global__ void k_loop_norm(float* __restrict__ la, const float* __restrict__ deg) {
  int i = blockIdx.x * blockDim.x + threadIdx.x;
  if (i >= N_ATOMS * EDIM) return;
  la[i] /= fmaxf(deg[i / EDIM], 1.f);
}

// --------------------------------------------------- type bucketing (hetero)
__global__ void k_count(const int* __restrict__ types, int* __restrict__ cnt) {
  int i = blockIdx.x * blockDim.x + threadIdx.x;
  if (i < N_ATOMS) atomicAdd(&cnt[types[i]], 1);
}
__global__ void k_offsets(const int* __restrict__ cnt, int* __restrict__ segOff,
                          int* __restrict__ cursor) {
  if (blockIdx.x == 0 && threadIdx.x == 0) {
    int acc = 0;
    for (int t = 0; t < TT; ++t) {
      segOff[t] = acc;
      acc += (cnt[t] + 15) & ~15;   // 16-align each type segment for WMMA tiles
      cursor[t] = 0;
    }
    segOff[TT] = acc;
  }
}
__global__ void k_scatter(const int* __restrict__ types, const int* __restrict__ segOff,
                          int* __restrict__ cursor, int* __restrict__ perm) {
  int i = blockIdx.x * blockDim.x + threadIdx.x;
  if (i >= N_ATOMS) return;
  int t = types[i];
  int pos = segOff[t] + atomicAdd(&cursor[t], 1);
  perm[pos] = i;
}

// ------------------------------------------------------------- WMMA GEMM
// Y[gr, 0..127] = act( X[gr, 0..K) @ W[type] + bias[type] )
// Block = 4 waves = one 16x128 output row-tile; each wave computes 16x32
// (two 16x16 WMMA tiles sharing one A fragment). The 16xKP A tile is staged
// into LDS once per block via GLOBAL_LOAD_ASYNC_TO_LDS_B128 (ASYNCcnt path,
// per-lane EXEC-masked so perm gather / padding lanes just drop out).
// perm==null -> identity rows, type 0.
__global__ void __launch_bounds__(128)
k_gemm(const float* __restrict__ X, int ldx, int K, int KP,
       const f16* __restrict__ Wt, const float* __restrict__ bias,
       float* __restrict__ Y, int ldy, int act,
       const int* __restrict__ perm, const int* __restrict__ segOff) {
  __shared__ float sA[16 * 160];          // KP <= 160
  int lane = threadIdx.x & 31;
  int warp = threadIdx.x >> 5;
  int rowBase = blockIdx.x * 16;

  // ---- stage 16 x KP block of A into LDS
  int kvec = KP >> 2;
  for (int idx = threadIdx.x; idx < 16 * kvec; idx += 128) {
    int r = idx / kvec;
    int c = (idx % kvec) * 4;
    int row = rowBase + r;
    int gr = perm ? perm[row] : row;
    float* dst = &sA[r * KP + c];
    if (gr >= 0 && c < K) {
      // direct global -> LDS async copy (no VGPR round-trip)
      unsigned ldsa = (unsigned)(size_t)dst;     // ISA 10.2: addr[31:0] == LDS offset
      unsigned long long ga = (unsigned long long)(size_t)(X + (size_t)gr * ldx + c);
      asm volatile("global_load_async_to_lds_b128 %0, %1, off"
                   :: "v"(ldsa), "v"(ga) : "memory");
    } else {
      *(float4*)dst = make_float4(0.f, 0.f, 0.f, 0.f);   // pad rows/cols
    }
  }
  asm volatile("s_wait_asynccnt 0x0" ::: "memory");
  __syncthreads();

  // ---- per-tile weight type (segments are 16-aligned so a tile has one type)
  int t = 0;
  if (segOff) {
    #pragma unroll
    for (int i = 1; i < TT; ++i)
      if (rowBase >= segOff[i]) t = i;
  }

  int m = lane & 15, half = lane >> 4;
  const float* arow = &sA[m * KP];
  int kb0 = half * 16;       // lanes 0-15: K 0..15 ; lanes 16-31: 16..31
  int kb1 = half * 16 + 8;
  const f16* wbase = Wt + (size_t)t * CH * KP;
  const f16* wcol0 = wbase + (size_t)(warp * 32 + m) * KP + half * 16;
  const f16* wcol1 = wcol0 + (size_t)16 * KP;

  v8f acc0 = {0.f, 0.f, 0.f, 0.f, 0.f, 0.f, 0.f, 0.f};
  v8f acc1 = {0.f, 0.f, 0.f, 0.f, 0.f, 0.f, 0.f, 0.f};
  for (int k0 = 0; k0 < KP; k0 += 32) {
    v16h a, b0, b1;
    #pragma unroll
    for (int i = 0; i < 8; ++i) {
      a[i]     = (f16)arow[k0 + kb0 + i];
      a[i + 8] = (f16)arow[k0 + kb1 + i];
    }
    #pragma unroll
    for (int i = 0; i < 16; ++i) {
      b0[i] = wcol0[k0 + i];
      b1[i] = wcol1[k0 + i];
    }
    acc0 = __builtin_amdgcn_wmma_f32_16x16x32_f16(false, a, false, b0,
                                                  (short)0, acc0, false, false);
    acc1 = __builtin_amdgcn_wmma_f32_16x16x32_f16(false, a, false, b1,
                                                  (short)0, acc1, false, false);
  }

  int n0 = warp * 32 + m;
  float bv0 = bias[(size_t)t * CH + n0];
  float bv1 = bias[(size_t)t * CH + n0 + 16];
  int mstart = half * 8;     // D layout: lanes 0-15 rows 0..7, lanes 16-31 rows 8..15
  #pragma unroll
  for (int r = 0; r < 8; ++r) {
    int row = rowBase + mstart + r;
    int gr = perm ? perm[row] : row;
    if (gr < 0) continue;
    float* yr = Y + (size_t)gr * ldy;
    float v0 = acc0[r] + bv0;
    float v1 = acc1[r] + bv1;
    if (act) { v0 = fmaxf(v0, 0.f); v1 = fmaxf(v1, 0.f); }
    yr[n0]      = v0;
    yr[n0 + 16] = v1;
  }
}

// ------------------------------------------------------------- GAT edge ops
// wave per edge: z = lrelu(xl[s]+xr[d]+ea@We); logits = <z, att>; seg-max.
__global__ void k_edge_logits(const int* __restrict__ ei, const float* __restrict__ ea,
                              const float* __restrict__ la,
                              const float* __restrict__ xl, const float* __restrict__ xr,
                              const float* __restrict__ We, const float* __restrict__ att,
                              float* __restrict__ logits, float* __restrict__ lmax) {
  int wid = (blockIdx.x * blockDim.x + threadIdx.x) >> 5;
  int lane = threadIdx.x & 31;
  if (wid >= ETOT) return;
  int s, d;
  const float* eav;
  if (wid < N_EDGES) { s = ei[wid]; d = ei[N_EDGES + wid]; eav = ea + (size_t)wid * EDIM; }
  else               { s = d = wid - N_EDGES;              eav = la + (size_t)(wid - N_EDGES) * EDIM; }
  float e0[EDIM];
  #pragma unroll
  for (int j = 0; j < EDIM; ++j) e0[j] = eav[j];
  float p0 = 0.f, p1 = 0.f;
  #pragma unroll
  for (int jj = 0; jj < 4; ++jj) {
    int c = lane + jj * 32;
    float ee = 0.f;
    #pragma unroll
    for (int j = 0; j < EDIM; ++j) ee += e0[j] * We[j * CH + c];
    float z = xl[(size_t)s * CH + c] + xr[(size_t)d * CH + c] + ee;
    z = (z > 0.f) ? z : 0.2f * z;
    float w = att[c] * z;
    if (c < 64) p0 += w; else p1 += w;
  }
  #pragma unroll
  for (int off = 16; off > 0; off >>= 1) {
    p0 += __shfl_xor(p0, off);
    p1 += __shfl_xor(p1, off);
  }
  if (lane == 0) {
    logits[(size_t)wid * 2]     = p0;
    logits[(size_t)wid * 2 + 1] = p1;
    atomicMaxF(&lmax[(size_t)d * 2],     p0);
    atomicMaxF(&lmax[(size_t)d * 2 + 1], p1);
  }
}

__global__ void k_edge_exp(const int* __restrict__ ei, const float* __restrict__ logits,
                           const float* __restrict__ lmax, float* __restrict__ exb,
                           float* __restrict__ denom) {
  int i = blockIdx.x * blockDim.x + threadIdx.x;
  if (i >= ETOT * 2) return;
  int e = i >> 1, h = i & 1;
  int d = (e < N_EDGES) ? ei[N_EDGES + e] : e - N_EDGES;
  float ex = __expf(logits[i] - lmax[(size_t)d * 2 + h]);
  exb[i] = ex;
  atomicAdd(&denom[(size_t)d * 2 + h], ex);
}

__global__ void k_edge_agg(const int* __restrict__ ei, const float* __restrict__ xl,
                           const float* __restrict__ exb, const float* __restrict__ denom,
                           float* __restrict__ out) {
  int wid = (blockIdx.x * blockDim.x + threadIdx.x) >> 5;
  int lane = threadIdx.x & 31;
  if (wid >= ETOT) return;
  int s, d;
  if (wid < N_EDGES) { s = ei[wid]; d = ei[N_EDGES + wid]; }
  else               { s = d = wid - N_EDGES; }
  float a0 = exb[(size_t)wid * 2]     / (denom[(size_t)d * 2]     + 1e-16f);
  float a1 = exb[(size_t)wid * 2 + 1] / (denom[(size_t)d * 2 + 1] + 1e-16f);
  #pragma unroll
  for (int jj = 0; jj < 4; ++jj) {
    int c = lane + jj * 32;
    float a = (c < 64) ? a0 : a1;
    atomicAdd(&out[(size_t)d * CH + c], a * xl[(size_t)s * CH + c]);
  }
}

__global__ void k_add_bias(float* __restrict__ buf, const float* __restrict__ bias) {
  int i = blockIdx.x * blockDim.x + threadIdx.x;
  if (i >= N_ATOMS * CH) return;
  buf[i] += bias[i & (CH - 1)];
}

__global__ void k_concat(const float* __restrict__ g, const float* __restrict__ x,
                         float* __restrict__ cat) {
  int i = blockIdx.x * blockDim.x + threadIdx.x;
  if (i >= N_ATOMS * (CH + INF_)) return;
  int r = i / (CH + INF_), c = i % (CH + INF_);
  cat[i] = (c < CH) ? g[(size_t)r * CH + c] : x[(size_t)r * INF_ + (c - CH)];
}

// ---------------------------------------------------------------- launcher
extern "C" void kernel_launch(void* const* d_in, const int* in_sizes, int n_in,
                              void* d_out, int out_size, void* d_ws, size_t ws_size,
                              hipStream_t stream) {
  const float* x      = (const float*)d_in[0];
  const int*   ei     = (const int*)d_in[1];
  const float* ea     = (const float*)d_in[2];
  const int*   types  = (const int*)d_in[3];
  const float* emb_W  = (const float*)d_in[4];
  const float* emb_b  = (const float*)d_in[5];
  const float* sh_W   = (const float*)d_in[6];
  const float* sh_b   = (const float*)d_in[7];
  const float* mu1_W  = (const float*)d_in[8];
  const float* mu1_b  = (const float*)d_in[9];
  const float* mu2_W  = (const float*)d_in[10];
  const float* mu2_b  = (const float*)d_in[11];
  const float* lv1_W  = (const float*)d_in[12];
  const float* lv1_b  = (const float*)d_in[13];
  const float* lv2_W  = (const float*)d_in[14];
  const float* lv2_b  = (const float*)d_in[15];
  // g1 / gmu / glv param blocks: Wl, bl, Wr, br, We, att, bias
  const float* G[3][7];
  for (int g = 0; g < 3; ++g)
    for (int p = 0; p < 7; ++p)
      G[g][p] = (const float*)d_in[16 + g * 7 + p];

  // ---- carve workspace
  char* wp = (char*)d_ws;
  auto alloc = [&](size_t bytes) -> void* {
    void* r = (void*)wp;
    wp += (bytes + 255) & ~(size_t)255;
    return r;
  };
  const size_t NF = (size_t)N_ATOMS * CH * sizeof(float);
  float* fA    = (float*)alloc(NF);                          // y / reused as h1
  float* fXL   = (float*)alloc(NF);
  float* fXR   = (float*)alloc(NF);
  float* fG    = (float*)alloc(NF);                          // GAT output
  float* fY2   = (float*)alloc(NF);                          // after shared hetero
  float* fCAT  = (float*)alloc((size_t)N_ATOMS * (CH + INF_) * sizeof(float));
  float* la    = (float*)alloc((size_t)N_ATOMS * EDIM * sizeof(float));
  float* deg   = (float*)alloc((size_t)N_ATOMS * sizeof(float));
  float* logits= (float*)alloc((size_t)ETOT * 2 * sizeof(float));
  float* exb   = (float*)alloc((size_t)ETOT * 2 * sizeof(float));
  float* lmax  = (float*)alloc((size_t)N_ATOMS * 2 * sizeof(float));
  float* denom = (float*)alloc((size_t)N_ATOMS * 2 * sizeof(float));
  int*   perm  = (int*)alloc((size_t)(N_ATOMS + 16 * TT) * sizeof(int));
  int*   cnt   = (int*)alloc(TT * sizeof(int));
  int*   segOff= (int*)alloc((TT + 1) * sizeof(int));
  int*   cursor= (int*)alloc(TT * sizeof(int));
  f16* embWt = (f16*)alloc((size_t)CH * 32 * sizeof(f16));
  f16* gWt[3][2];
  for (int g = 0; g < 3; ++g)
    for (int p = 0; p < 2; ++p)
      gWt[g][p] = (f16*)alloc((size_t)CH * CH * sizeof(f16));
  f16* shWt  = (f16*)alloc((size_t)TT * CH * CH * sizeof(f16));
  f16* mu1Wt = (f16*)alloc((size_t)TT * CH * 160 * sizeof(f16));
  f16* mu2Wt = (f16*)alloc((size_t)TT * CH * CH * sizeof(f16));
  f16* lv1Wt = (f16*)alloc((size_t)TT * CH * 160 * sizeof(f16));
  f16* lv2Wt = (f16*)alloc((size_t)TT * CH * CH * sizeof(f16));

  const int B = 256;
  auto fillf = [&](float* p, float v, int n) {
    k_fill_f32<<<cdiv(n, B), B, 0, stream>>>(p, v, n);
  };

  // ---- weight prep (f32 -> f16, transpose, pad K)
  auto prep = [&](const float* W, f16* Wt, int K, int KP, int Tn) {
    int total = Tn * CH * KP;
    k_prep_w<<<cdiv(total, B), B, 0, stream>>>(W, Wt, K, KP, Tn);
  };
  prep(emb_W, embWt, INF_, 32, 1);
  for (int g = 0; g < 3; ++g) {
    prep(G[g][0], gWt[g][0], CH, CH, 1);   // Wl
    prep(G[g][2], gWt[g][1], CH, CH, 1);   // Wr
  }
  prep(sh_W,  shWt,  CH,        CH,  TT);
  prep(mu1_W, mu1Wt, CH + INF_, 160, TT);
  prep(mu2_W, mu2Wt, CH,        CH,  TT);
  prep(lv1_W, lv1Wt, CH + INF_, 160, TT);
  prep(lv2_W, lv2Wt, CH,        CH,  TT);

  // ---- self-loop attr = mean of incoming edge_attr
  fillf(deg, 0.f, N_ATOMS);
  fillf(la, 0.f, N_ATOMS * EDIM);
  k_loop_accum<<<cdiv(N_EDGES * EDIM, B), B, 0, stream>>>(ei, ea, deg, la);
  k_loop_norm<<<cdiv(N_ATOMS * EDIM, B), B, 0, stream>>>(la, deg);

  // ---- type bucketing for hetero layers (built once, reused 5x)
  k_fill_i32<<<cdiv(N_ATOMS + 16 * TT, B), B, 0, stream>>>(perm, -1, N_ATOMS + 16 * TT);
  k_fill_i32<<<1, 32, 0, stream>>>(cnt, 0, TT);
  k_count<<<cdiv(N_ATOMS, B), B, 0, stream>>>(types, cnt);
  k_offsets<<<1, 32, 0, stream>>>(cnt, segOff, cursor);
  k_scatter<<<cdiv(N_ATOMS, B), B, 0, stream>>>(types, segOff, cursor, perm);

  auto gemm = [&](const float* X, int ldx, int K, int KP, const f16* Wt,
                  const float* bias, float* Y, int ldy, int act,
                  const int* pp, const int* so) {
    int rowTiles = pp ? (N_ATOMS + 16 * TT) / 16 : N_ATOMS / 16;
    k_gemm<<<dim3(rowTiles), 128, 0, stream>>>(X, ldx, K, KP, Wt, bias, Y, ldy,
                                               act, pp, so);
  };

  auto gat = [&](int g, const float* xin, float* out) {
    gemm(xin, CH, CH, CH, gWt[g][0], G[g][1], fXL, CH, 0, nullptr, nullptr);
    gemm(xin, CH, CH, CH, gWt[g][1], G[g][3], fXR, CH, 0, nullptr, nullptr);
    fillf(lmax, -INFINITY, N_ATOMS * 2);
    fillf(denom, 0.f, N_ATOMS * 2);
    fillf(out, 0.f, N_ATOMS * CH);
    int eblocks = cdiv(ETOT * 32, B);
    k_edge_logits<<<eblocks, B, 0, stream>>>(ei, ea, la, fXL, fXR,
                                             G[g][4], G[g][5], logits, lmax);
    k_edge_exp<<<cdiv(ETOT * 2, B), B, 0, stream>>>(ei, logits, lmax, exb, denom);
    k_edge_agg<<<eblocks, B, 0, stream>>>(ei, fXL, exb, denom, out);
    k_add_bias<<<cdiv(N_ATOMS * CH, B), B, 0, stream>>>(out, G[g][6]);
  };

  // ---- pipeline
  // y = relu(x @ emb_W + emb_b)
  gemm(x, INF_, INF_, 32, embWt, emb_b, fA, CH, 1, nullptr, nullptr);
  // GAT 1
  gat(0, fA, fG);
  // shared hetero
  gemm(fG, CH, CH, CH, shWt, sh_b, fY2, CH, 0, perm, segOff);
  // mu branch
  gat(1, fY2, fG);
  k_concat<<<cdiv(N_ATOMS * (CH + INF_), B), B, 0, stream>>>(fG, x, fCAT);
  gemm(fCAT, CH + INF_, CH + INF_, 160, mu1Wt, mu1_b, fA, CH, 0, perm, segOff);
  gemm(fA, CH, CH, CH, mu2Wt, mu2_b, (float*)d_out, CH, 0, perm, segOff);
  // lv branch
  gat(2, fY2, fG);
  k_concat<<<cdiv(N_ATOMS * (CH + INF_), B), B, 0, stream>>>(fG, x, fCAT);
  gemm(fCAT, CH + INF_, CH + INF_, 160, lv1Wt, lv1_b, fA, CH, 0, perm, segOff);
  gemm(fA, CH, CH, CH, lv2Wt, lv2_b, (float*)d_out + (size_t)N_ATOMS * CH, CH, 0,
       perm, segOff);
}
